// LSTM_VAE_20684562498320
// MI455X (gfx1250) — matrix operand order repaired and checked by reading, not compiled, not run on hardware
//
#include <hip/hip_runtime.h>
#include <hip/hip_bf16.h>

typedef __attribute__((ext_vector_type(16))) _Float16 v16h;
typedef __attribute__((ext_vector_type(8)))  _Float16 v8h;
typedef __attribute__((ext_vector_type(8)))  float    v8f;

#define WMMA_F16(a, b, c) \
  __builtin_amdgcn_wmma_f32_16x16x32_f16(false, (a), false, (b), (short)0, (c), false, false)

// ---------------------------------------------------------------------------
// WMMA operand gathers (wave32, shapes per CDNA5 ISA §7.12.2)
// ---------------------------------------------------------------------------

// A-matrix 16x32 f16 from row-major (lda halfs). lanes 0-15: M=lane, K={kb..kb+7, kb+16..kb+23};
// lanes 16-31: same M, K offset +8.  Two contiguous 16B chunks per lane.
__device__ __forceinline__ v16h load_A(const _Float16* base, int lda, int kb) {
  const int lane = threadIdx.x & 31;
  const int m    = lane & 15;
  const int koff = (lane & 16) ? 8 : 0;
  const _Float16* p = base + m * lda + kb + koff;
  v8h lo = *(const v8h*)(p);
  v8h hi = *(const v8h*)(p + 16);
  return __builtin_shufflevector(lo, hi, 0,1,2,3,4,5,6,7,8,9,10,11,12,13,14,15);
}

// B-matrix 32x16 f16, B[k][n] = W[n][k] with W row-major (ldw halfs).
// lanes 0-15: N=n0+lane, K={kb..kb+7, kb+16..kb+23} ; lanes 16-31: N=n0+lane-16, K offset +8.
// Two contiguous 8-half chunks per lane -> two b128 loads, no conversion VALU.
__device__ __forceinline__ v16h load_Bh(const _Float16* W, int ldw, int n0, int kb) {
  const int lane = threadIdx.x & 31;
  const int n    = n0 + (lane & 15);
  const int koff = (lane & 16) ? 8 : 0;
  const _Float16* p = W + n * ldw + kb + koff;
  v8h lo = *(const v8h*)(p);
  v8h hi = *(const v8h*)(p + 16);
  return __builtin_shufflevector(lo, hi, 0,1,2,3,4,5,6,7,8,9,10,11,12,13,14,15);
}

__device__ __forceinline__ float sigmoidf_fast(float x) {
  return 1.f / (1.f + __expf(-x));
}

// f16 weight-arena offsets (in halfs), all 16B aligned
#define WO_EW1   0        // 64x64
#define WO_EW2   4096     // 64x64
#define WO_EW3   8192     // 32x64
#define WO_DW1P  10240    // 64x32 (K padded 16->32 with zeros)
#define WO_DW2   12288    // 64x64
#define WO_DW3   16384    // 128x64
#define WO_TOTAL 24576

// ---------------------------------------------------------------------------
// Kernel 1a: f32 -> f16 conversion of X and W_ih into workspace
// ---------------------------------------------------------------------------
__global__ void k_convert(const float* __restrict__ x, const float* __restrict__ wih,
                          _Float16* __restrict__ xh, _Float16* __restrict__ wih_h,
                          int nx, int nw) {
  int i = blockIdx.x * blockDim.x + threadIdx.x;
  int stride = gridDim.x * blockDim.x;
  for (int k = i; k < nx; k += stride) xh[k] = (_Float16)x[k];
  for (int k = i; k < nw; k += stride) wih_h[k] = (_Float16)wih[k];
}

// ---------------------------------------------------------------------------
// Kernel 1b: MLP weights -> f16 arena (done once; arena stays L2-resident).
// dW1 (64x16) is zero-padded to 64x32 so the K=16 layer uses the plain path.
// ---------------------------------------------------------------------------
__global__ void k_convert_w(const float* __restrict__ eW1, const float* __restrict__ eW2,
                            const float* __restrict__ eW3, const float* __restrict__ dW1,
                            const float* __restrict__ dW2, const float* __restrict__ dW3,
                            _Float16* __restrict__ Wh) {
  int i = blockIdx.x * blockDim.x + threadIdx.x;
  int stride = gridDim.x * blockDim.x;
  for (int k = i; k < 4096; k += stride) Wh[WO_EW1 + k] = (_Float16)eW1[k];
  for (int k = i; k < 4096; k += stride) Wh[WO_EW2 + k] = (_Float16)eW2[k];
  for (int k = i; k < 2048; k += stride) Wh[WO_EW3 + k] = (_Float16)eW3[k];
  for (int k = i; k < 2048; k += stride) {
    int n = k >> 5, kk = k & 31;
    Wh[WO_DW1P + k] = (kk < 16) ? (_Float16)dW1[n * 16 + kk] : (_Float16)0.f;
  }
  for (int k = i; k < 4096; k += stride) Wh[WO_DW2 + k] = (_Float16)dW2[k];
  for (int k = i; k < 8192; k += stride) Wh[WO_DW3 + k] = (_Float16)dW3[k];
}

// ---------------------------------------------------------------------------
// Kernel 2: Gx[t, 0:256] = X[t,:] @ W_ih^T + b_ih + b_hh   (WMMA f16, f32 accum)
// One block = 16 timesteps; 8 waves, each wave owns 2 N-tiles of 16 gates.
// ---------------------------------------------------------------------------
__global__ __launch_bounds__(256) void k_gemm_ih(const _Float16* __restrict__ Xh,
                                                 const _Float16* __restrict__ Wih,
                                                 const float* __restrict__ b_ih,
                                                 const float* __restrict__ b_hh,
                                                 float* __restrict__ Gx) {
  const int t0   = blockIdx.x * 16;
  const int wave = threadIdx.x >> 5;
  const int lane = threadIdx.x & 31;
  const int n16  = lane & 15;
  const int mo   = (lane & 16) ? 8 : 0;
  const _Float16* At = Xh + (size_t)t0 * 128;
#pragma unroll
  for (int w = 0; w < 2; ++w) {
    const int n0 = (wave * 2 + w) * 16;
    v8f c = {};
#pragma unroll
    for (int kb = 0; kb < 128; kb += 32) {
      v16h a = load_A(At, 128, kb);
      v16h b = load_Bh(Wih, 128, n0, kb);
      c = WMMA_F16(a, b, c);
    }
    const int n = n0 + n16;
    const float bias = b_ih[n] + b_hh[n];
#pragma unroll
    for (int r = 0; r < 8; ++r)
      Gx[(size_t)(t0 + mo + r) * 256 + n] = c[r] + bias;
  }
}

// ---------------------------------------------------------------------------
// Kernel 3: sequential LSTM scan (the serial latency chain).
// One workgroup, 256 threads: thread j owns gate j (W_hh row j in VGPRs).
// h lives in LDS (broadcast reads); c lives in registers of threads 0..63.
// ---------------------------------------------------------------------------
__global__ __launch_bounds__(256) void k_scan(const float* __restrict__ Gx,
                                              const float* __restrict__ W_hh,
                                              const float* __restrict__ h0,
                                              const float* __restrict__ c0,
                                              _Float16* __restrict__ hs_h, int T) {
  __shared__ __align__(16) float sh[64];
  __shared__ __align__(16) float sg[256];
  const int j = threadIdx.x;

  float w[64];
#pragma unroll
  for (int k = 0; k < 64; ++k) w[k] = W_hh[j * 64 + k];

  float c = 0.f;
  if (j < 64) { sh[j] = h0[j]; c = c0[j]; }
  __syncthreads();

  for (int t = 0; t < T; ++t) {
    // keep the Gx stream ahead of the chain (global_prefetch_b8)
    int tp = t + 8; if (tp >= T) tp = T - 1;
    __builtin_prefetch(&Gx[(size_t)tp * 256 + j], 0, 0);

    float acc = Gx[(size_t)t * 256 + j];
#pragma unroll
    for (int k = 0; k < 64; ++k) acc += w[k] * sh[k];
    sg[j] = acc;
    __syncthreads();

    if (j < 64) {
      const float gi = sg[j];
      const float gf = sg[64 + j];
      const float gg = sg[128 + j];
      const float go = sg[192 + j];
      c = sigmoidf_fast(gf) * c + sigmoidf_fast(gi) * tanhf(gg);
      const float h = sigmoidf_fast(go) * tanhf(c);
      sh[j] = h;
      hs_h[(size_t)t * 64 + j] = (_Float16)h;
    }
    __syncthreads();
  }
}

// ---------------------------------------------------------------------------
// Kernel 4: fused VAE head. One wave per 16-timestep tile; the whole
// e1 -> e2 -> (mu,logvar) -> z -> d1 -> d2 -> out chain stays on-chip,
// with LDS ping-pong buffers to re-layout WMMA D tiles into A tiles.
// All weights come from the f16 arena (two b128 loads per B tile, no cvt).
// ---------------------------------------------------------------------------
__global__ __launch_bounds__(32) void k_mlp(
    const _Float16* __restrict__ hs_h, const _Float16* __restrict__ Wh,
    const float* __restrict__ eb1, const float* __restrict__ ea1,
    const float* __restrict__ eb2, const float* __restrict__ ea2,
    const float* __restrict__ eb3,
    const float* __restrict__ db1, const float* __restrict__ da1,
    const float* __restrict__ db2, const float* __restrict__ da2,
    const float* __restrict__ db3,
    const float* __restrict__ eps,
    float* __restrict__ out_p, float* __restrict__ mu_p, float* __restrict__ lv_p) {
  __shared__ __align__(16) _Float16 B1[16 * 64];
  __shared__ __align__(16) _Float16 B2[16 * 64];

  const int t0   = blockIdx.x * 16;
  const int lane = threadIdx.x & 31;
  const int n16  = lane & 15;
  const int mo   = (lane & 16) ? 8 : 0;
  const float a1 = ea1[0], a2 = ea2[0], a3 = da1[0], a4 = da2[0];

  // ---- e1 = prelu(hs @ eW1^T + eb1) : A read straight from global f16 ----
#pragma unroll
  for (int nt = 0; nt < 4; ++nt) {
    const int n0 = nt * 16;
    v8f c = {};
#pragma unroll
    for (int kb = 0; kb < 64; kb += 32)
      c = WMMA_F16(load_A(hs_h + (size_t)t0 * 64, 64, kb), load_Bh(Wh + WO_EW1, 64, n0, kb), c);
    const float bias = eb1[n0 + n16];
#pragma unroll
    for (int r = 0; r < 8; ++r) {
      float v = c[r] + bias;
      B1[(mo + r) * 64 + n0 + n16] = (_Float16)(v >= 0.f ? v : a1 * v);
    }
  }
  __syncthreads();

  // ---- e2 = prelu(e1 @ eW2^T + eb2) ----
#pragma unroll
  for (int nt = 0; nt < 4; ++nt) {
    const int n0 = nt * 16;
    v8f c = {};
#pragma unroll
    for (int kb = 0; kb < 64; kb += 32)
      c = WMMA_F16(load_A(B1, 64, kb), load_Bh(Wh + WO_EW2, 64, n0, kb), c);
    const float bias = eb2[n0 + n16];
#pragma unroll
    for (int r = 0; r < 8; ++r) {
      float v = c[r] + bias;
      B2[(mo + r) * 64 + n0 + n16] = (_Float16)(v >= 0.f ? v : a2 * v);
    }
  }
  __syncthreads();

  // ---- ml = e2 @ eW3^T + eb3 ; mu | logvar ; z = mu + eps*exp(0.5*logvar) ----
  v8f cmu = {}, clv = {};
#pragma unroll
  for (int kb = 0; kb < 64; kb += 32) {
    v16h a = load_A(B2, 64, kb);
    cmu = WMMA_F16(a, load_Bh(Wh + WO_EW3, 64, 0, kb),  cmu);
    clv = WMMA_F16(a, load_Bh(Wh + WO_EW3, 64, 16, kb), clv);
  }
  {
    const float bm = eb3[n16], bl = eb3[16 + n16];
#pragma unroll
    for (int r = 0; r < 8; ++r) {
      const int t = t0 + mo + r;
      const float mu = cmu[r] + bm;
      const float lv = clv[r] + bl;
      mu_p[(size_t)t * 16 + n16] = mu;
      lv_p[(size_t)t * 16 + n16] = lv;
      const float z = mu + eps[(size_t)t * 16 + n16] * __expf(0.5f * lv);
      B1[(mo + r) * 32 + n16]      = (_Float16)z;  // z tile 16x32, K 16..31 zero
      B1[(mo + r) * 32 + 16 + n16] = (_Float16)0.f;
    }
  }
  __syncthreads();

  // ---- d1 = prelu(z @ dW1^T + db1), K padded 16->32 in the weight arena ----
#pragma unroll
  for (int nt = 0; nt < 4; ++nt) {
    const int n0 = nt * 16;
    v8f c = WMMA_F16(load_A(B1, 32, 0), load_Bh(Wh + WO_DW1P, 32, n0, 0), (v8f){});
    const float bias = db1[n0 + n16];
#pragma unroll
    for (int r = 0; r < 8; ++r) {
      float v = c[r] + bias;
      B2[(mo + r) * 64 + n0 + n16] = (_Float16)(v >= 0.f ? v : a3 * v);
    }
  }
  __syncthreads();

  // ---- d2 = prelu(d1 @ dW2^T + db2) ----
#pragma unroll
  for (int nt = 0; nt < 4; ++nt) {
    const int n0 = nt * 16;
    v8f c = {};
#pragma unroll
    for (int kb = 0; kb < 64; kb += 32)
      c = WMMA_F16(load_A(B2, 64, kb), load_Bh(Wh + WO_DW2, 64, n0, kb), c);
    const float bias = db2[n0 + n16];
#pragma unroll
    for (int r = 0; r < 8; ++r) {
      float v = c[r] + bias;
      B1[(mo + r) * 64 + n0 + n16] = (_Float16)(v >= 0.f ? v : a4 * v);
    }
  }
  __syncthreads();

  // ---- out = tanh(d2 @ dW3^T + db3), 128 outputs = 8 N-tiles ----
#pragma unroll
  for (int nt = 0; nt < 8; ++nt) {
    const int n0 = nt * 16;
    v8f c = {};
#pragma unroll
    for (int kb = 0; kb < 64; kb += 32)
      c = WMMA_F16(load_A(B1, 64, kb), load_Bh(Wh + WO_DW3, 64, n0, kb), c);
    const float bias = db3[n0 + n16];
#pragma unroll
    for (int r = 0; r < 8; ++r)
      out_p[(size_t)(t0 + mo + r) * 128 + n0 + n16] = tanhf(c[r] + bias);
  }
}

// ---------------------------------------------------------------------------
// Launch
// ---------------------------------------------------------------------------
extern "C" void kernel_launch(void* const* d_in, const int* in_sizes, int n_in,
                              void* d_out, int out_size, void* d_ws, size_t ws_size,
                              hipStream_t stream) {
  const float* x    = (const float*)d_in[0];
  const float* W_ih = (const float*)d_in[1];
  const float* W_hh = (const float*)d_in[2];
  const float* b_ih = (const float*)d_in[3];
  const float* b_hh = (const float*)d_in[4];
  const float* h0   = (const float*)d_in[5];
  const float* c0   = (const float*)d_in[6];
  const float* eW1  = (const float*)d_in[7];
  const float* eb1  = (const float*)d_in[8];
  const float* ea1  = (const float*)d_in[9];
  const float* eW2  = (const float*)d_in[10];
  const float* eb2  = (const float*)d_in[11];
  const float* ea2  = (const float*)d_in[12];
  const float* eW3  = (const float*)d_in[13];
  const float* eb3  = (const float*)d_in[14];
  const float* dW1  = (const float*)d_in[15];
  const float* db1  = (const float*)d_in[16];
  const float* da1  = (const float*)d_in[17];
  const float* dW2  = (const float*)d_in[18];
  const float* db2  = (const float*)d_in[19];
  const float* da2  = (const float*)d_in[20];
  const float* dW3  = (const float*)d_in[21];
  const float* db3  = (const float*)d_in[22];
  const float* eps  = (const float*)d_in[23];

  const int T = in_sizes[0] / 128;   // 131072

  // workspace carve-up (all offsets 256B aligned)
  char* ws = (char*)d_ws;
  _Float16* Xh    = (_Float16*)ws;                                   // T*128 f16
  _Float16* Wih_h = (_Float16*)(ws + (size_t)T * 128 * 2);           // 256*128 f16
  float*    Gx    = (float*)(ws + (size_t)T * 128 * 2 + 65536);      // T*256 f32
  _Float16* hs_h  = (_Float16*)((char*)Gx + (size_t)T * 256 * 4);    // T*64  f16
  _Float16* Wh    = hs_h + (size_t)T * 64;                           // WO_TOTAL f16

  float* out_p = (float*)d_out;
  float* mu_p  = out_p + (size_t)T * 128;
  float* lv_p  = mu_p + (size_t)T * 16;

  k_convert<<<1024, 256, 0, stream>>>(x, W_ih, Xh, Wih_h, T * 128, 256 * 128);
  k_convert_w<<<64, 256, 0, stream>>>(eW1, eW2, eW3, dW1, dW2, dW3, Wh);
  k_gemm_ih<<<T / 16, 256, 0, stream>>>(Xh, Wih_h, b_ih, b_hh, Gx);
  k_scan<<<1, 256, 0, stream>>>(Gx, W_hh, h0, c0, hs_h, T);
  k_mlp<<<T / 16, 32, 0, stream>>>(hs_h, Wh,
                                   eb1, ea1, eb2, ea2, eb3,
                                   db1, da1, db2, da2, db3,
                                   eps, out_p, mu_p, lv_p);
}
